// _Upsample_67224828117164
// MI455X (gfx1250) — matrix operand order repaired and checked by reading, not compiled, not run on hardware
//
#include <hip/hip_runtime.h>

typedef _Float16 v16h __attribute__((ext_vector_type(16)));
typedef float    v8f  __attribute__((ext_vector_type(8)));

#define C_CH   512
#define B_N    8
#define KTOT   2048      // 512 cin * 4 folded taps
#define XP_H   34        // 32 + 2 pad rows
#define XP_W   40        // 32 + pad, padded stride
#define LDS_ST 48        // f16 stride per LDS row (96B: 32B-aligned fragments)

// K-permutation within each 32-group so the WMMA A fragment
// (lanes 0-15: K=0..7,16..23 ; lanes 16-31: K=8..15,24..31 per ISA 7.12.2)
// becomes one contiguous 16-f16 (32B) read: swap bits 3 and 4.
__device__ __forceinline__ int permk(int kk) {
    return (kk & 7) | ((kk & 8) << 1) | ((kk & 16) >> 1);
}

// Raw 32-bit LDS byte offset for async-DMA VDST operand.
__device__ __forceinline__ unsigned lds_off(const void* p) {
    return (unsigned)(size_t)(__attribute__((address_space(3))) const void*)p;
}

// GLOBAL_LOAD_ASYNC_TO_LDS_B128: 16B per lane, ASYNCcnt-tracked (ISA 08 §4).
__device__ __forceinline__ void async_b128(const _Float16* g, const _Float16* l) {
    asm volatile("global_load_async_to_lds_b128 %0, %1, off"
                 :: "v"(lds_off(l)), "v"((unsigned long long)(size_t)g)
                 : "memory");
}

__device__ __forceinline__ void async_wait0() {
    asm volatile("s_wait_asynccnt 0" ::: "memory");
}

// ---------------------------------------------------------------------------
// Kernel 1: fold 3x3 weights into 4 parity-specific 2x2 f16 kernels.
// Layout: wf[par][cout][ group*32 + permk(k) ], k = cin*4 + (ty*2+tx)
// ---------------------------------------------------------------------------
__global__ void fold_weights(const float* __restrict__ w, _Float16* __restrict__ wf) {
    int idx = blockIdx.x * blockDim.x + threadIdx.x;      // 4*512*512 threads
    if (idx >= 4 * C_CH * C_CH) return;
    const int par = idx >> 18;
    const int rem = idx & ((1 << 18) - 1);
    const int co  = rem >> 9;
    const int ci  = rem & 511;
    const int a = par >> 1, p = par & 1;

    const float* wp = w + ((size_t)co * C_CH + ci) * 9;
    float r[3][2];
#pragma unroll
    for (int ky = 0; ky < 3; ++ky) {
        const float w0 = wp[ky * 3 + 0], w1 = wp[ky * 3 + 1], w2 = wp[ky * 3 + 2];
        r[ky][0] = p ? (w0 + w1) : w0;
        r[ky][1] = p ? w2 : (w1 + w2);
    }
    float f[2][2];
#pragma unroll
    for (int tx = 0; tx < 2; ++tx) {
        f[0][tx] = a ? (r[0][tx] + r[1][tx]) : r[0][tx];
        f[1][tx] = a ? r[2][tx] : (r[1][tx] + r[2][tx]);
    }
    const size_t base = ((size_t)par * C_CH + co) * KTOT;
#pragma unroll
    for (int ty = 0; ty < 2; ++ty)
#pragma unroll
        for (int tx = 0; tx < 2; ++tx) {
            const int klin = ci * 4 + ty * 2 + tx;
            wf[base + (klin & ~31) + permk(klin & 31)] = (_Float16)f[ty][tx];
        }
}

// ---------------------------------------------------------------------------
// Kernel 2: fp32 x -> zero-padded f16  xp[b][c][34][40]
// ---------------------------------------------------------------------------
__global__ void pad_x(const float* __restrict__ x, _Float16* __restrict__ xp, int total) {
    int idx = blockIdx.x * blockDim.x + threadIdx.x;
    if (idx >= total) return;
    const int pc = idx % XP_W;
    const int t  = idx / XP_W;
    const int pr = t % XP_H;
    const int bc = t / XP_H;     // b*512 + c
    float v = 0.0f;
    if (pr >= 1 && pr <= 32 && pc >= 1 && pc <= 32)
        v = x[((size_t)bc * 32 + (pr - 1)) * 32 + (pc - 1)];
    xp[idx] = (_Float16)v;
}

// ---------------------------------------------------------------------------
// Kernel 3: implicit-GEMM upsample-conv, double-buffered, async A-tile DMA.
// Grid: 1024 blocks = 8 batch * 4 parity * 4 mtile * 8 ntile, 256 threads.
// Block tile: 128 cout x 128 spatial, K = 2048, WMMA f32_16x16x32_f16.
// ---------------------------------------------------------------------------
__global__ void __launch_bounds__(256)
upconv_wmma(const _Float16* __restrict__ wf, const _Float16* __restrict__ xp,
            const float* __restrict__ bias, float* __restrict__ out) {
    __shared__ __align__(32) _Float16 ldsA[2][128 * LDS_ST];
    __shared__ __align__(32) _Float16 ldsB[2][128 * LDS_ST];

    const int tid  = threadIdx.x;
    const int lane = tid & 31;
    const int wave = tid >> 5;
    const int wm = wave & 3;          // 4 waves along M (32 rows each)
    const int wn = wave >> 2;         // 2 waves along N (64 cols each)
    const int lm = lane & 15;
    const int kh = lane >> 4;         // K-half select per ISA fragment layout

    int bid = blockIdx.x;
    const int nt  = bid & 7;  bid >>= 3;
    const int mt  = bid & 3;  bid >>= 2;
    const int par = bid & 3;  bid >>= 2;
    const int b   = bid;
    const int a = par >> 1, p = par & 1;

    // A-tile async DMA: 8KB tile = 512 x 16B chunks; 2 chunks per thread.
    const int c0 = tid, c1 = tid + 256;
    const int r0 = c0 >> 2, q0 = c0 & 3;
    const int r1 = c1 >> 2, q1 = c1 & 3;
    const _Float16* gA0 = wf + ((size_t)(par * C_CH + mt * 128 + r0)) * KTOT + q0 * 8;
    const _Float16* gA1 = wf + ((size_t)(par * C_CH + mt * 128 + r1)) * KTOT + q1 * 8;
    const int lA0 = r0 * LDS_ST + q0 * 8;
    const int lA1 = r1 * LDS_ST + q1 * 8;

    // B-tile gather: thread owns one k column, 16 contiguous spatial points
    const int bk = tid >> 3;                 // 0..31 : k within chunk
    const int bn = (tid & 7) * 16;           // n offset within 128-tile
    const int n0 = nt * 128 + bn;
    const int h0 = n0 >> 5;
    const int w0 = n0 & 31;
    const int bOff = bn * LDS_ST + permk(bk);

    v8f acc[2][4];
#pragma unroll
    for (int f = 0; f < 2; ++f)
#pragma unroll
        for (int g = 0; g < 4; ++g) acc[f][g] = 0;

    _Float16 breg[16];
    // gather B chunk 'kb' into breg
#define GATHER_B(kbase_)                                                              \
    {                                                                                 \
        const int klin = (kbase_) + bk;                                               \
        const int cin  = klin >> 2;                                                   \
        const int tap  = klin & 3;                                                    \
        const _Float16* bs = xp +                                                     \
            ((size_t)(b * C_CH + cin) * XP_H + (h0 + a + (tap >> 1))) * XP_W          \
            + (w0 + p + (tap & 1));                                                   \
        _Pragma("unroll")                                                             \
        for (int j = 0; j < 16; ++j) breg[j] = bs[j];                                 \
    }

    // ---- prologue: fill buffer 0
    async_b128(gA0, &ldsA[0][lA0]);
    async_b128(gA1, &ldsA[0][lA1]);
    GATHER_B(0);
    async_wait0();
#pragma unroll
    for (int j = 0; j < 16; ++j) ldsB[0][bOff + j * LDS_ST] = breg[j];
    __syncthreads();

    for (int kb = 0; kb < 64; ++kb) {
        const int cur = kb & 1, nxt = cur ^ 1;

        if (kb < 63) {
            const int kn = (kb + 1) * 32;
            async_b128(gA0 + kn, &ldsA[nxt][lA0]);   // DMA overlaps WMMA phase
            async_b128(gA1 + kn, &ldsA[nxt][lA1]);
            GATHER_B(kn);
        }

        v16h af[2], bf[4];
#pragma unroll
        for (int f = 0; f < 2; ++f)
            af[f] = *(const v16h*)&ldsA[cur][(wm * 32 + f * 16 + lm) * LDS_ST + kh * 16];
#pragma unroll
        for (int g = 0; g < 4; ++g)
            bf[g] = *(const v16h*)&ldsB[cur][(wn * 64 + g * 16 + lm) * LDS_ST + kh * 16];

#pragma unroll
        for (int f = 0; f < 2; ++f)
#pragma unroll
            for (int g = 0; g < 4; ++g)
                acc[f][g] = __builtin_amdgcn_wmma_f32_16x16x32_f16(
                    false, af[f], false, bf[g], (short)0, acc[f][g], false, false);

        if (kb < 63) {
            async_wait0();                           // our A-chunks landed in ldsA[nxt]
#pragma unroll
            for (int j = 0; j < 16; ++j) ldsB[nxt][bOff + j * LDS_ST] = breg[j];
        }
        __syncthreads();                             // publish nxt, retire cur
    }

    // Epilogue: D layout per ISA 7.12.2 (VGPR v: M = v + 8*(lane>>4), N = lane&15)
#pragma unroll
    for (int f = 0; f < 2; ++f) {
        const int cobase = mt * 128 + wm * 32 + f * 16 + kh * 8;
#pragma unroll
        for (int g = 0; g < 4; ++g) {
            const int ng = nt * 128 + wn * 64 + g * 16 + lm;
            const int h = ng >> 5, w = ng & 31;
            const size_t sp = (size_t)(2 * h + a) * 64 + (2 * w + p);
#pragma unroll
            for (int v = 0; v < 8; ++v) {
                const int co = cobase + v;
                out[(((size_t)b * C_CH + co) << 12) + sp] = acc[f][g][v] + bias[co];
            }
        }
    }
#undef GATHER_B
}

// ---------------------------------------------------------------------------
extern "C" void kernel_launch(void* const* d_in, const int* in_sizes, int n_in,
                              void* d_out, int out_size, void* d_ws, size_t ws_size,
                              hipStream_t stream) {
    const float* x    = (const float*)d_in[0];
    const float* w    = (const float*)d_in[1];
    const float* bias = (const float*)d_in[2];
    float* out = (float*)d_out;

    _Float16* wfold = (_Float16*)d_ws;                                       // 8 MB
    _Float16* xpad  = (_Float16*)((char*)d_ws + (size_t)4 * C_CH * KTOT * 2);

    fold_weights<<<(4 * C_CH * C_CH + 255) / 256, 256, 0, stream>>>(w, wfold);

    const int xp_total = B_N * C_CH * XP_H * XP_W;
    pad_x<<<(xp_total + 255) / 256, 256, 0, stream>>>(x, xpad, xp_total);

    upconv_wmma<<<B_N * 4 * 4 * 8, 256, 0, stream>>>(wfold, xpad, bias, out);
}
